// UnionRGATLayer2_12180527251909
// MI455X (gfx1250) — compile-verified, matrix operand order checked by reading, not run on hardware
//
#include <hip/hip_runtime.h>
#include <hip/hip_fp16.h>

#define N_NODES 50000
#define N_EDGES 600000
#define DDIM    128
#define N_RELS  460

typedef _Float16 v16h __attribute__((ext_vector_type(16)));
typedef _Float16 v8h  __attribute__((ext_vector_type(8)));
typedef float    v8f  __attribute__((ext_vector_type(8)));
typedef unsigned int v4u __attribute__((ext_vector_type(4)));
typedef int      v8i  __attribute__((ext_vector_type(8)));
typedef int      v4i  __attribute__((ext_vector_type(4)));

// ---------- helpers ----------
__device__ __forceinline__ void atomAddF(float* p, float v) {
    __hip_atomic_fetch_add(p, v, __ATOMIC_RELAXED, __HIP_MEMORY_SCOPE_AGENT);
}

__device__ __forceinline__ void atomMaxF(float* addr, float val) {
    if (val >= 0.0f) atomicMax((int*)addr, __float_as_int(val));
    else             atomicMin((unsigned int*)addr, __float_as_uint(val));
}

// ---------- 0: zero output, init softmax state ----------
__global__ __launch_bounds__(256) void k_init(float* out, float* m, float* den) {
    int i = blockIdx.x * 256 + threadIdx.x;
    if (i < N_NODES * DDIM) out[i] = 0.0f;
    if (i < N_NODES) { m[i] = -__builtin_inff(); den[i] = 0.0f; }
}

// ---------- 1: u = attn_fc_w.T @ attn_fc2_w ----------
__global__ void k_compute_u(const float* __restrict__ W,   // [D, 3D] row-major
                            const float* __restrict__ w2,  // [D]
                            float* __restrict__ u) {       // [3D]
    int k = threadIdx.x;            // 0..383
    float acc = 0.0f;
    for (int o = 0; o < DDIM; ++o) acc += w2[o] * W[o * (3 * DDIM) + k];
    u[k] = acc;
}

// ---------- 2: per-node / per-relation scalar scores (one wave per item) ----------
__global__ __launch_bounds__(256) void k_scores(const float* __restrict__ h,
                                                const float* __restrict__ emb_rel,
                                                const float* __restrict__ u,
                                                float* __restrict__ s_src,
                                                float* __restrict__ s_dst,
                                                float* __restrict__ s_rel) {
    int gw   = (blockIdx.x * 256 + threadIdx.x) >> 5;
    int lane = threadIdx.x & 31;
    if (gw < N_NODES) {
        float4 hv = *(const float4*)(h + (size_t)gw * DDIM + lane * 4);
        float4 u0 = *(const float4*)(u + lane * 4);
        float4 u1 = *(const float4*)(u + DDIM + lane * 4);
        float a = hv.x * u0.x + hv.y * u0.y + hv.z * u0.z + hv.w * u0.w;
        float b = hv.x * u1.x + hv.y * u1.y + hv.z * u1.z + hv.w * u1.w;
        #pragma unroll
        for (int off = 16; off > 0; off >>= 1) {
            a += __shfl_xor(a, off, 32);
            b += __shfl_xor(b, off, 32);
        }
        if (lane == 0) { s_src[gw] = a; s_dst[gw] = b; }
    } else if (gw < N_NODES + N_RELS) {
        int r = gw - N_NODES;
        float4 rv = *(const float4*)(emb_rel + (size_t)r * DDIM + lane * 4);
        float4 u2 = *(const float4*)(u + 2 * DDIM + lane * 4);
        float a = rv.x * u2.x + rv.y * u2.y + rv.z * u2.z + rv.w * u2.w;
        #pragma unroll
        for (int off = 16; off > 0; off >>= 1) a += __shfl_xor(a, off, 32);
        if (lane == 0) s_rel[r] = a;
    }
}

// ---------- 3: edge logits (leaky_relu) + segment max ----------
__global__ __launch_bounds__(256) void k_edge_logit(const int* __restrict__ esrc,
                                                    const int* __restrict__ edst,
                                                    const int* __restrict__ etyp,
                                                    const float* __restrict__ s_src,
                                                    const float* __restrict__ s_dst,
                                                    const float* __restrict__ s_rel,
                                                    float* __restrict__ eW,
                                                    float* __restrict__ m) {
    int i = blockIdx.x * 256 + threadIdx.x;
    if (i >= N_EDGES) return;
    float x = s_src[esrc[i]] + s_dst[edst[i]] + s_rel[etyp[i]];
    float e = x > 0.0f ? x : 0.01f * x;
    eW[i] = e;
    atomMaxF(m + edst[i], e);
}

// ---------- 4: exp(e - m[dst]) + segment sum ----------
__global__ __launch_bounds__(256) void k_edge_exp(const int* __restrict__ edst,
                                                  const float* __restrict__ m,
                                                  float* __restrict__ eW,
                                                  float* __restrict__ den) {
    int i = blockIdx.x * 256 + threadIdx.x;
    if (i >= N_EDGES) return;
    int d = edst[i];
    float ex = expf(eW[i] - m[d]);
    eW[i] = ex;
    atomAddF(den + d, ex);
}

// ---------- 5: alpha-weighted scatter of src features (one wave per edge) ----------
__global__ __launch_bounds__(256) void k_edge_scatter(const int* __restrict__ esrc,
                                                      const int* __restrict__ edst,
                                                      const float* __restrict__ eW,
                                                      const float* __restrict__ den,
                                                      const float* __restrict__ h,
                                                      float* __restrict__ out) {
    int gw   = (blockIdx.x * 256 + threadIdx.x) >> 5;
    int lane = threadIdx.x & 31;
    if (gw >= N_EDGES) return;
    int s = esrc[gw], d = edst[gw];
    float alpha = eW[gw] / den[d];
    float4 hv = *(const float4*)(h + (size_t)s * DDIM + lane * 4);
    float* o = out + (size_t)d * DDIM + lane * 4;
    atomAddF(o + 0, alpha * hv.x);
    atomAddF(o + 1, alpha * hv.y);
    atomAddF(o + 2, alpha * hv.z);
    atomAddF(o + 3, alpha * hv.w);
}

// ---------- 6: self-loop GEMM  out += h @ loop_weight ----------
// loop_weight staged into LDS via TDM (tensor_load_to_lds), then transposed +
// converted to f16 so WMMA B fragments are contiguous 16B ds_load_b128 reads.
__global__ __launch_bounds__(256) void k_selfloop_wmma(const float* __restrict__ h,
                                                       const float* __restrict__ w,
                                                       float* __restrict__ out) {
    __shared__ float    wraw[DDIM * DDIM];   // 64 KB: raw f32 loop_weight
    __shared__ _Float16 wT[DDIM * DDIM];     // 32 KB: f16, transposed [n][k]

#if __has_builtin(__builtin_amdgcn_tensor_load_to_lds)
    if (threadIdx.x == 0) {
        unsigned long long ga = (unsigned long long)(uintptr_t)w;
        // ISA 10.2: LDS aperture maps flat addr[31:0] directly to LDS offset
        unsigned lds_off = (unsigned)(uintptr_t)&wraw[0];
        // D# group0: count=1 | lds_addr | global_addr[56:0] | type=2
        v4u g0 = { 1u, lds_off, (unsigned)ga,
                   (unsigned)((ga >> 32) & 0x1FFFFFFu) | (2u << 30) };
        // D# group1: data_size=4B(2); tensor_dim0=128, tensor_dim1=128;
        //            tile_dim0=128, tile_dim1=128; tensor_dim0_stride=128
        v8i g1 = { (int)(2u << 16),            // workgroup_mask=0, data_size=2
                   (int)(128u << 16),          // tensor_dim0 lo16 in [63:48]
                   (int)(0u | (128u << 16)),   // tensor_dim0 hi16, tensor_dim1 lo16
                   (int)(0u | (128u << 16)),   // tensor_dim1 hi16, tile_dim0
                   (int)(128u),                // tile_dim1, tile_dim2=0
                   (int)(128u),                // tensor_dim0_stride lo32
                   0, 0 };                     // stride hi, dim1_stride (2D)
        v4i gz4 = { 0, 0, 0, 0 };              // groups 2/3: unused (2D tile)
        v8i gz8 = { 0, 0, 0, 0, 0, 0, 0, 0 };  // extra group (6-arg toolchain)
        __builtin_amdgcn_tensor_load_to_lds(g0, g1, gz4, gz4, gz8, 0);
        __builtin_amdgcn_s_wait_tensorcnt(0);
    }
    __syncthreads();
#else
    for (int idx = threadIdx.x; idx < DDIM * DDIM; idx += 256) wraw[idx] = w[idx];
    __syncthreads();
#endif

    // transpose + f32->f16 convert: wT[n][k] = (f16) w[k][n]
    for (int idx = threadIdx.x; idx < DDIM * DDIM; idx += 256) {
        int k = idx >> 7, n = idx & 127;
        wT[n * DDIM + k] = (_Float16)wraw[idx];
    }
    __syncthreads();

    int wave = threadIdx.x >> 5;
    int lane = threadIdx.x & 31;
    int tile = blockIdx.x * 8 + wave;         // 3125*8 = 25000 tiles
    int rowt = tile >> 3, colt = tile & 7;
    int row0 = rowt * 16, col0 = colt * 16;
    int hi = (lane >= 16) ? 1 : 0;
    int mn = lane & 15;

    // C/D layout: VGPR r -> M = r + 8*hi, N = lane&15  (accumulate onto scatter result)
    v8f c;
    #pragma unroll
    for (int r = 0; r < 8; ++r)
        c[r] = out[(row0 + r + hi * 8) * DDIM + col0 + mn];

    const float* arow = h + (row0 + mn) * DDIM;
    const _Float16* brow = wT + (col0 + mn) * DDIM;

    #pragma unroll
    for (int kk = 0; kk < DDIM; kk += 32) {
        // 16-bit frag layout: elem j=8g+q -> K = 16g + 8*hi + q, q=0..7 contiguous
        v16h a, b;
        v8h b0 = *(const v8h*)(brow + kk + hi * 8);        // ds_load_b128
        v8h b1 = *(const v8h*)(brow + kk + 16 + hi * 8);   // ds_load_b128
        #pragma unroll
        for (int q = 0; q < 8; ++q) {
            a[q]     = (_Float16)arow[kk + hi * 8 + q];
            a[8 + q] = (_Float16)arow[kk + 16 + hi * 8 + q];
            b[q]     = b0[q];
            b[8 + q] = b1[q];
        }
        c = __builtin_amdgcn_wmma_f32_16x16x32_f16(
                false, a, false, b, (short)0, c, false, false);
    }

    #pragma unroll
    for (int r = 0; r < 8; ++r)
        out[(row0 + r + hi * 8) * DDIM + col0 + mn] = c[r];
}

// ---------- host launcher ----------
extern "C" void kernel_launch(void* const* d_in, const int* in_sizes, int n_in,
                              void* d_out, int out_size, void* d_ws, size_t ws_size,
                              hipStream_t stream) {
    const float* h        = (const float*)d_in[0];
    const float* emb_rel  = (const float*)d_in[1];
    const float* attn_fc  = (const float*)d_in[2];   // [D, 3D]
    const float* attn_fc2 = (const float*)d_in[3];   // [1, D]
    const float* loop_w   = (const float*)d_in[4];   // [D, D]
    const int*   esrc     = (const int*)d_in[5];
    const int*   edst     = (const int*)d_in[6];
    const int*   etyp     = (const int*)d_in[7];
    float* out = (float*)d_out;

    float* ws    = (float*)d_ws;
    float* u     = ws;
    float* s_src = u + 3 * DDIM;
    float* s_dst = s_src + N_NODES;
    float* s_rel = s_dst + N_NODES;
    float* m     = s_rel + N_RELS;
    float* den   = m + N_NODES;
    float* eW    = den + N_NODES;

    constexpr int initBlocks  = (N_NODES * DDIM + 255) / 256;
    constexpr int scoreBlocks = ((N_NODES + N_RELS) * 32 + 255) / 256;
    constexpr int edgeBlocks  = (N_EDGES + 255) / 256;
    constexpr int scatBlocks  = (N_EDGES * 32 + 255) / 256;
    constexpr int gemmBlocks  = (N_NODES / 16) * (DDIM / 16) / 8;

    k_init<<<initBlocks, 256, 0, stream>>>(out, m, den);
    k_compute_u<<<1, 3 * DDIM, 0, stream>>>(attn_fc, attn_fc2, u);
    k_scores<<<scoreBlocks, 256, 0, stream>>>(h, emb_rel, u, s_src, s_dst, s_rel);
    k_edge_logit<<<edgeBlocks, 256, 0, stream>>>(esrc, edst, etyp, s_src, s_dst, s_rel, eW, m);
    k_edge_exp<<<edgeBlocks, 256, 0, stream>>>(edst, m, eW, den);
    k_edge_scatter<<<scatBlocks, 256, 0, stream>>>(esrc, edst, eW, den, h, out);
    k_selfloop_wmma<<<gemmBlocks, 256, 0, stream>>>(h, loop_w, out);
}
